// HeteroGNN_57303453663976
// MI455X (gfx1250) — matrix operand order repaired and checked by reading, not compiled, not run on hardware
//
#include <hip/hip_runtime.h>

typedef __attribute__((ext_vector_type(16))) __bf16 v16bf;
typedef __attribute__((ext_vector_type(8)))  float  v8f;

#define NU 50000
#define NS 50000
#define DIN 128
#define DHID 256
#define DOUT 128

// ---------------- zero fill ----------------
__global__ void k_zero(float* __restrict__ p, size_t n) {
  size_t i = (size_t)blockIdx.x * blockDim.x + threadIdx.x;
  size_t st = (size_t)gridDim.x * blockDim.x;
  for (; i < n; i += st) p[i] = 0.0f;
}

// ---------------- scatter sum (+ optional degree count), one wave per edge ----------------
__global__ void k_scatter(const float* __restrict__ xsrc,
                          const int* __restrict__ srci,
                          const int* __restrict__ dsti,
                          float* __restrict__ agg,
                          float* __restrict__ cnt,
                          int nE, int D) {
  int w = blockIdx.x * (blockDim.x >> 5) + (threadIdx.x >> 5);
  if (w >= nE) return;                      // wave-uniform exit
  int lane = threadIdx.x & 31;
  int s = srci[w];
  int d = dsti[w];
  const float4* sp = (const float4*)(xsrc + (size_t)s * D);
  float* dp = agg + (size_t)d * D;
  int nvec = D >> 2;
  for (int v = lane; v < nvec; v += 32) {
    float4 val = sp[v];
    atomicAdd(dp + 4 * v + 0, val.x);
    atomicAdd(dp + 4 * v + 1, val.y);
    atomicAdd(dp + 4 * v + 2, val.z);
    atomicAdd(dp + 4 * v + 3, val.w);
  }
  if (cnt != nullptr && lane == 0) atomicAdd(cnt + d, 1.0f);
}

// ---------------- pack weights into WMMA bf16 B fragments ----------------
// Combined B (K x N): rows 0..K1-1 = Wl^T (Wl is [Nout,K1] row-major),
// rows K1..K1+K2-1 = Wr^T (Wr is [Nout,K2]).
// B 32x16 layout (matches A): lanes 0-15 hold K {0..7} then {16..23}, lanes 16-31 hold K {8..15} then {24..31}.
// Storage: frag slot index = ((kc*NTtot + nt)*32 + lane), 16 bf16 each.
__global__ void k_packB(const float* __restrict__ Wl,
                        const float* __restrict__ Wr,
                        __bf16* __restrict__ out,
                        int K1, int K2, int Nout) {
  int t = blockIdx.x * blockDim.x + threadIdx.x;
  int KC = (K1 + K2) >> 5;
  int NT = Nout >> 4;
  int lane = t & 31;
  int kc = (t >> 5) % KC;
  int nt = (t >> 5) / KC;
  if (nt >= NT) return;
  int kh = lane >> 4;
  int r  = lane & 15;
  int n  = nt * 16 + r;
  __bf16* dst = out + (((size_t)kc * NT + nt) * 32 + lane) * 16;
  for (int e = 0; e < 16; ++e) {
    int k = kc * 32 + (e < 8 ? kh * 8 + e : 16 + kh * 8 + (e - 8));
    float wv = (k < K1) ? Wl[(size_t)n * K1 + k]
                        : Wr[(size_t)n * K2 + (k - K1)];
    dst[e] = (__bf16)wv;
  }
}

// ---------------- build 16x32 bf16 A fragment from an fp32 row ----------------
// ISA 16-bit A 16x32 layout: lanes 0-15 (M=lane): K {0..7} then {16..23};
// lanes 16-31 (M=lane-16): K {8..15} then {24..31}.  scale folds the mean divide.
__device__ __forceinline__ v16bf build_afrag(const float* __restrict__ rowp,
                                             int kbase, float scale, int kh) {
  const float* p0 = rowp + kbase + kh * 8;
  const float* p1 = rowp + kbase + 16 + kh * 8;
  float4 f0 = *(const float4*)(p0);
  float4 f1 = *(const float4*)(p0 + 4);
  float4 f2 = *(const float4*)(p1);
  float4 f3 = *(const float4*)(p1 + 4);
  v16bf a;
  a[0]  = (__bf16)(f0.x * scale); a[1]  = (__bf16)(f0.y * scale);
  a[2]  = (__bf16)(f0.z * scale); a[3]  = (__bf16)(f0.w * scale);
  a[4]  = (__bf16)(f1.x * scale); a[5]  = (__bf16)(f1.y * scale);
  a[6]  = (__bf16)(f1.z * scale); a[7]  = (__bf16)(f1.w * scale);
  a[8]  = (__bf16)(f2.x * scale); a[9]  = (__bf16)(f2.y * scale);
  a[10] = (__bf16)(f2.z * scale); a[11] = (__bf16)(f2.w * scale);
  a[12] = (__bf16)(f3.x * scale); a[13] = (__bf16)(f3.y * scale);
  a[14] = (__bf16)(f3.z * scale); a[15] = (__bf16)(f3.w * scale);
  return a;
}

// ---------------- fused SAGE GEMM: out = act( (A1/cnt)@Wl^T + A2@Wr^T + bias ) ----------------
// One wave per 16-row M tile; fixed 8 N-tiles (128 cols) per wave -> 64 C VGPRs, no spills.
// blockIdx.y selects the 128-column group. B fragments staged through LDS per round
// (8 K-chunks = 64KB), shared by all 8 waves of the block.
__global__ __launch_bounds__(256)
void k_sage_gemm(const float* __restrict__ A1,
                 const float* __restrict__ cnt,
                 const float* __restrict__ A2,
                 const __bf16* __restrict__ Bp,
                 const float* __restrict__ bias,
                 float* __restrict__ out,
                 int M, int K1, int K2, int Nout, int dorelu) {
  __shared__ __bf16 sB[8 * 8 * 32 * 16];   // 64KB: up to 8 staged K-chunks
  const int NTtot = Nout >> 4;
  const int ntb   = blockIdx.y << 3;        // base 16-col tile of this column group
  const int tid   = threadIdx.x;
  const int mt    = blockIdx.x * (blockDim.x >> 5) + (tid >> 5);
  const int mtiles = M >> 4;
  const bool active = (mt < mtiles);        // wave-uniform; inactive waves still barrier
  const int lane = tid & 31;
  const int kh = lane >> 4;
  const int r  = lane & 15;
  const int row = (active ? mt : 0) * 16 + r;

  float invc = 1.0f;
  if (active && cnt != nullptr) invc = 1.0f / fmaxf(cnt[row], 1.0f);

  v8f z8 = {0.f, 0.f, 0.f, 0.f, 0.f, 0.f, 0.f, 0.f};
  v8f c[8];
  #pragma unroll
  for (int nt = 0; nt < 8; ++nt) c[nt] = z8;

  const int KC1   = K1 >> 5;
  const int KCtot = (K1 + K2) >> 5;
  const float* row1 = A1 + (size_t)row * K1;
  const float* row2 = (A2 != nullptr) ? (A2 + (size_t)row * K2) : nullptr;

  for (int kc0 = 0; kc0 < KCtot; kc0 += 8) {
    const int kcr = (KCtot - kc0 < 8) ? (KCtot - kc0) : 8;
    // cooperative stage: kcr chunks x 8 nt x 32 lane-slots of 32B each
    for (int cch = tid; cch < kcr * 8 * 32; cch += 256) {
      int kk  = cch >> 8;                  // / (8*32)
      int rem = cch & 255;
      int nt  = rem >> 5;
      int ls  = rem & 31;
      size_t g = (((size_t)(kc0 + kk) * NTtot + ntb + nt) * 32 + ls) * 16;
      *(v16bf*)(sB + (size_t)cch * 16) = *(const v16bf*)(Bp + g);
    }
    __syncthreads();
    if (active) {
      for (int kk = 0; kk < kcr; ++kk) {
        int kcg = kc0 + kk;
        v16bf a;
        if (kcg < KC1) a = build_afrag(row1, kcg * 32, invc, kh);
        else           a = build_afrag(row2, (kcg - KC1) * 32, 1.0f, kh);
        #pragma unroll
        for (int nt = 0; nt < 8; ++nt) {
          v16bf b = *(const v16bf*)(sB + (((kk * 8 + nt) * 32) + lane) * 16);
          c[nt] = __builtin_amdgcn_wmma_f32_16x16x32_bf16(
              false, a, false, b, (short)0, c[nt], false, false);
        }
      }
    }
    __syncthreads();
  }

  if (active) {
    // C layout: VGPR j holds [M = j + kh*8, N = r] of each 16x16 tile
    #pragma unroll
    for (int nt = 0; nt < 8; ++nt) {
      int n = (ntb + nt) * 16 + r;
      float bv = bias[n];
      #pragma unroll
      for (int j = 0; j < 8; ++j) {
        float v = c[nt][j] + bv;
        if (dorelu) v = fmaxf(v, 0.0f);
        out[(size_t)(mt * 16 + kh * 8 + j) * Nout + n] = v;
      }
    }
  }
}

extern "C" void kernel_launch(void* const* d_in, const int* in_sizes, int n_in,
                              void* d_out, int out_size, void* d_ws, size_t ws_size,
                              hipStream_t stream) {
  const float* x_user  = (const float*)d_in[0];
  const float* x_song  = (const float*)d_in[1];
  const int*   e_us    = (const int*)d_in[2];
  const int*   e_su    = (const int*)d_in[3];
  const float* c1us_Wl = (const float*)d_in[4];
  const float* c1us_bl = (const float*)d_in[5];
  const float* c1us_Wr = (const float*)d_in[6];
  const float* c1su_Wl = (const float*)d_in[7];
  const float* c1su_bl = (const float*)d_in[8];
  const float* c1su_Wr = (const float*)d_in[9];
  const float* c2us_Wl = (const float*)d_in[10];
  const float* c2us_bl = (const float*)d_in[11];
  const float* c2us_Wr = (const float*)d_in[12];
  const float* c2su_Wl = (const float*)d_in[13];
  const float* c2su_bl = (const float*)d_in[14];
  const float* c2su_Wr = (const float*)d_in[15];
  const float* pu_W    = (const float*)d_in[16];
  const float* pu_b    = (const float*)d_in[17];
  const float* ps_W    = (const float*)d_in[18];
  const float* ps_b    = (const float*)d_in[19];

  const int nE = in_sizes[2] / 2;

  // -------- workspace carve-out (256B-aligned) --------
  size_t off = 0;
  auto fal = [&](size_t nfl) -> float* {
    float* p = (float*)((char*)d_ws + off);
    off += ((nfl * sizeof(float) + 255) & ~(size_t)255);
    return p;
  };
  float* agg1_s = fal((size_t)NS * DIN);
  float* agg1_u = fal((size_t)NU * DIN);
  float* cnt_s  = fal((size_t)NS);
  float* cnt_u  = fal((size_t)NU);
  float* h_s    = fal((size_t)NS * DHID);
  float* h_u    = fal((size_t)NU * DHID);
  float* agg2_s = fal((size_t)NS * DHID);
  float* agg2_u = fal((size_t)NU * DHID);
  auto bal = [&](size_t nbf) -> __bf16* {
    __bf16* p = (__bf16*)((char*)d_ws + off);
    off += ((nbf * 2 + 255) & ~(size_t)255);
    return p;
  };
  __bf16* Bp_c1us = bal(65536);   // KC=8,  NT=16
  __bf16* Bp_c1su = bal(65536);
  __bf16* Bp_c2us = bal(65536);   // KC=16, NT=8
  __bf16* Bp_c2su = bal(65536);
  __bf16* Bp_pu   = bal(16384);   // KC=4,  NT=8
  __bf16* Bp_ps   = bal(16384);
  // conv2 outputs overlay conv1 accumulators (dead after conv1 GEMMs)
  float* z_s = agg1_s;
  float* z_u = agg1_u;

  float* out_user = (float*)d_out;
  float* out_song = out_user + (size_t)NU * DOUT;

  // -------- zero accumulators --------
  k_zero<<<1024, 256, 0, stream>>>(agg1_s, (size_t)NS * DIN);
  k_zero<<<1024, 256, 0, stream>>>(agg1_u, (size_t)NU * DIN);
  k_zero<<<64,   256, 0, stream>>>(cnt_s,  (size_t)NS);
  k_zero<<<64,   256, 0, stream>>>(cnt_u,  (size_t)NU);
  k_zero<<<2048, 256, 0, stream>>>(agg2_s, (size_t)NS * DHID);
  k_zero<<<2048, 256, 0, stream>>>(agg2_u, (size_t)NU * DHID);

  // -------- pack weights to bf16 B fragments --------
  k_packB<<<16, 256, 0, stream>>>(c1us_Wl, c1us_Wr, Bp_c1us, DIN, DIN, DHID);
  k_packB<<<16, 256, 0, stream>>>(c1su_Wl, c1su_Wr, Bp_c1su, DIN, DIN, DHID);
  k_packB<<<16, 256, 0, stream>>>(c2us_Wl, c2us_Wr, Bp_c2us, DHID, DHID, DOUT);
  k_packB<<<16, 256, 0, stream>>>(c2su_Wl, c2su_Wr, Bp_c2su, DHID, DHID, DOUT);
  k_packB<<<4,  256, 0, stream>>>(pu_W,    pu_W,    Bp_pu,   DOUT, 0, DOUT);
  k_packB<<<4,  256, 0, stream>>>(ps_W,    ps_W,    Bp_ps,   DOUT, 0, DOUT);

  // -------- conv1: scatter then fused GEMM+bias+root+ReLU --------
  int sblocks = (nE + 7) / 8;               // 8 waves / 256-thread block
  k_scatter<<<sblocks, 256, 0, stream>>>(x_user, e_us, e_us + nE, agg1_s, cnt_s, nE, DIN);
  k_scatter<<<sblocks, 256, 0, stream>>>(x_song, e_su, e_su + nE, agg1_u, cnt_u, nE, DIN);

  int gblocks = (NS / 16 + 7) / 8;          // 3125 M-tiles, 8 waves / block
  dim3 g2(gblocks, 2);                      // Nout=256 -> two 128-col groups
  dim3 g1(gblocks, 1);
  k_sage_gemm<<<g2, 256, 0, stream>>>(agg1_s, cnt_s, x_song, Bp_c1us, c1us_bl, h_s, NS, DIN, DIN, DHID, 1);
  k_sage_gemm<<<g2, 256, 0, stream>>>(agg1_u, cnt_u, x_user, Bp_c1su, c1su_bl, h_u, NU, DIN, DIN, DHID, 1);

  // -------- conv2 (degree counts reused: same edge sets) --------
  k_scatter<<<sblocks, 256, 0, stream>>>(h_u, e_us, e_us + nE, agg2_s, nullptr, nE, DHID);
  k_scatter<<<sblocks, 256, 0, stream>>>(h_s, e_su, e_su + nE, agg2_u, nullptr, nE, DHID);

  k_sage_gemm<<<g1, 256, 0, stream>>>(agg2_s, cnt_s, h_s, Bp_c2us, c2us_bl, z_s, NS, DHID, DHID, DOUT, 1);
  k_sage_gemm<<<g1, 256, 0, stream>>>(agg2_u, cnt_u, h_u, Bp_c2su, c2su_bl, z_u, NU, DHID, DHID, DOUT, 1);

  // -------- per-type post linear (no relu, no mean, no root) --------
  k_sage_gemm<<<g1, 256, 0, stream>>>(z_u, nullptr, nullptr, Bp_pu, pu_b, out_user, NU, DOUT, 0, DOUT, 0);
  k_sage_gemm<<<g1, 256, 0, stream>>>(z_s, nullptr, nullptr, Bp_ps, ps_b, out_song, NS, DOUT, 0, DOUT, 0);

  (void)n_in; (void)out_size; (void)ws_size;
}
